// UNetWithScharr_22995254903037
// MI455X (gfx1250) — matrix operand order repaired and checked by reading, not compiled, not run on hardware
//
#include <hip/hip_runtime.h>

// =====================================================================
// UNet-with-Scharr forward for MI455X (gfx1250).
// conv3x3 as implicit GEMM on V_WMMA_F32_16X16X32_F16 (f16 in, f32 acc),
// LDS double-buffered with software-pipelined global->reg->LDS staging,
// division-free incremental addressing.
// =====================================================================

typedef __attribute__((ext_vector_type(16))) _Float16 v16h;
typedef __attribute__((ext_vector_type(8)))  _Float16 v8h;
typedef __attribute__((ext_vector_type(8)))  float    v8f;
typedef __attribute__((ext_vector_type(4)))  float    v4f;

#define TM    64    // Cout tile
#define TN    128   // pixel tile
#define KSTEP 32    // K per WMMA step
#define LSTR  40    // LDS row stride in halves (80B rows, 16B aligned)

// ---------------------------------------------------------------------
// conv3x3 SAME via implicit GEMM.
//  pin : f16 zero-padded NHWC [2, H+2, W+2, Cin]
//  wts : f16 [Cout, Kp], k = (ky*3+kx)*Cin + ci, Kp = roundup32(9*Cin)
//  out : f32 NHWC [2, H, W, Cout]
// 256 threads = 8 wave32. Wave w: M-subtile (w&3), N-subtiles 4*(w>>2)+0..3
// Fast path (Cin%32==0): each K-step is one (ky,kx) tap, contiguous ci ->
// pointer-increment addressing, 1 barrier/step, and a 3-phase pipeline:
//   barrier -> issue next-tile global loads -> WMMA current tile -> ds_store
// so the HBM latency of the next tile is hidden under the current WMMAs.
// ---------------------------------------------------------------------
__global__ __launch_bounds__(256)
void conv3x3_wmma_kernel(const _Float16* __restrict__ pin,
                         const _Float16* __restrict__ wts,
                         float* __restrict__ out,
                         int H, int W, int Cin, int Cout)
{
    __shared__ _Float16 As[2][TM * LSTR];   // weights  [m][k]
    __shared__ _Float16 Bs[2][TN * LSTR];   // activations [pixel][k]

    const int tid  = threadIdx.x;
    const int lane = tid & 31;
    const int wave = tid >> 5;

    const int K  = 9 * Cin;
    const int Kp = (K + 31) & ~31;

    const int tileP = blockIdx.x;           // 128-pixel tile
    const int tileM = blockIdx.y;           // 64-Cout tile

    const int HW = H * W;
    const int Wp = W + 2;

    // cooperative-load assignments
    const int rowA = tid >> 2;              // 0..63
    const int kgA  = (tid & 3) * 8;         // 0,8,16,24
    const int rowB = tid >> 1;              // 0..127
    const int kgB  = (tid & 1) * 16;        // 0,16

    // this thread's B-row pixel coordinates
    int pB = tileP * TN + rowB;
    int bB = pB / HW;
    int qB = pB - bB * HW;
    int yB = qB / W;
    int xB = qB - yB * W;
    const _Float16* baseB = pin + (size_t)bB * (H + 2) * Wp * Cin;

    v8f acc[4] = {{}, {}, {}, {}};
    const int mi = wave & 3;                // M subtile
    const int nb = (wave >> 2) * 4;         // first of 4 N subtiles

    const int half = lane >> 4;
    const int lrow = lane & 15;

    // fragment compute from buffer `buf`: preload A + all 4 B fragments,
    // then issue the 4 WMMAs back-to-back (single dscnt wait).
    auto compute = [&](int buf) {
        v16h afrag;
        {
            const _Float16* ap = &As[buf][(mi * 16 + lrow) * LSTR + half * 8];
            v8h lo = *(const v8h*)ap;
            v8h hi = *(const v8h*)(ap + 16);
            #pragma unroll
            for (int i = 0; i < 8; ++i) { afrag[i] = lo[i]; afrag[i + 8] = hi[i]; }
        }
        v16h bfrag[4];
        #pragma unroll
        for (int t = 0; t < 4; ++t) {
            const _Float16* bp = &Bs[buf][((nb + t) * 16 + lrow) * LSTR + half * 8];
            v8h lo  = *(const v8h*)bp;
            v8h hi8 = *(const v8h*)(bp + 16);
            #pragma unroll
            for (int i = 0; i < 8; ++i) { bfrag[t][i] = lo[i]; bfrag[t][i + 8] = hi8[i]; }
        }
        #pragma unroll
        for (int t = 0; t < 4; ++t)
            acc[t] = __builtin_amdgcn_wmma_f32_16x16x32_f16(
                false, afrag, false, bfrag[t], (short)0, acc[t], false, false);
    };

    if ((Cin & 31) == 0) {
        // ---------------- fast path: division-free, double buffered --------
        const int cinSteps = Cin >> 5;      // K-steps per (ky,kx) tap
        const int nsteps   = 9 * cinSteps;

        const _Float16* asrc = wts + (size_t)(tileM * TM + rowA) * Kp + kgA;
        const _Float16* bsrc = baseB + ((size_t)yB * Wp + xB) * Cin + kgB;
        int scnt = 0, kxc = 0;

        auto advance = [&]() {
            asrc += KSTEP;
            bsrc += KSTEP;
            if (++scnt == cinSteps) {       // next (ky,kx) tap
                scnt = 0;
                if (kxc == 2) { kxc = 0; bsrc += (size_t)(Wp - 3) * Cin; }
                else ++kxc;
            }
        };

        // prologue: stage step 0 directly into buffer 0
        *(v8h*)&As[0][rowA * LSTR + kgA]     = *(const v8h*)asrc;
        *(v8h*)&Bs[0][rowB * LSTR + kgB]     = *(const v8h*)bsrc;
        *(v8h*)&Bs[0][rowB * LSTR + kgB + 8] = *(const v8h*)(bsrc + 8);
        advance();

        int buf = 0;
        for (int step = 0; step < nsteps; ++step) {
            __syncthreads();                // buf ready; buf^1 fully consumed
            v8h gA = {}, gB0 = {}, gB1 = {};
            const bool hasNext = (step + 1 < nsteps);
            if (hasNext) {                  // issue next-tile loads (no wait yet)
                gA  = *(const v8h*)asrc;
                gB0 = *(const v8h*)bsrc;
                gB1 = *(const v8h*)(bsrc + 8);
                advance();
                __builtin_prefetch(asrc, 0, 1);   // global_prefetch_b8
            }
            compute(buf);                   // WMMAs overlap the global loads
            if (hasNext) {                  // loadcnt wait lands here, post-WMMA
                int nxt = buf ^ 1;
                *(v8h*)&As[nxt][rowA * LSTR + kgA]     = gA;
                *(v8h*)&Bs[nxt][rowB * LSTR + kgB]     = gB0;
                *(v8h*)&Bs[nxt][rowB * LSTR + kgB + 8] = gB1;
            }
            buf ^= 1;
        }
    } else {
        // ---------------- fallback (Cin==2 stem): div-based gather ---------
        for (int k0 = 0; k0 < Kp; k0 += KSTEP) {
            __syncthreads();
            *(v8h*)&As[0][rowA * LSTR + kgA] =
                *(const v8h*)(wts + (size_t)(tileM * TM + rowA) * Kp + k0 + kgA);
            for (int j = 0; j < 16; ++j) {
                int kk = k0 + kgB + j;
                _Float16 v = (_Float16)0.f;
                if (kk < K) {
                    int ci = kk % Cin;
                    int r  = kk / Cin;
                    int ky = r / 3;
                    int kx = r - ky * 3;
                    v = baseB[((size_t)(yB + ky) * Wp + (xB + kx)) * Cin + ci];
                }
                Bs[0][rowB * LSTR + kgB + j] = v;
            }
            __syncthreads();
            compute(0);
        }
    }

    // ---- epilogue: C layout VGPR r -> m = r + 8*(lane/16), n = lane%16 ----
    {
        const int co0 = tileM * TM + mi * 16 + half * 8;
        #pragma unroll
        for (int t = 0; t < 4; ++t) {
            int p = tileP * TN + (nb + t) * 16 + lrow;
            float* dst = out + (size_t)p * Cout + co0;
            v4f lo4, hi4;
            #pragma unroll
            for (int i = 0; i < 4; ++i) { lo4[i] = acc[t][i]; hi4[i] = acc[t][i + 4]; }
            *(v4f*)dst       = lo4;
            *(v4f*)(dst + 4) = hi4;
        }
    }
}

// ---------------------------------------------------------------------
// Weight repack: f32 OIHW [Cout,Cin,3,3] -> f16 [Cout,Kp], zero-padded K
// ---------------------------------------------------------------------
__global__ void convert_weights_kernel(const float* __restrict__ w,
                                       _Float16* __restrict__ dst,
                                       int Cout, int Cin)
{
    int K = 9 * Cin, Kp = (K + 31) & ~31;
    int total = Cout * Kp;
    for (int i = blockIdx.x * blockDim.x + threadIdx.x; i < total;
         i += gridDim.x * blockDim.x) {
        int m = i / Kp;
        int k = i - m * Kp;
        _Float16 v = (_Float16)0.f;
        if (k < K) {
            int ci = k % Cin;
            int r  = k / Cin;
            int ky = r / 3;
            int kx = r - ky * 3;
            v = (_Float16)w[(((size_t)m * Cin + ci) * 3 + ky) * 3 + kx];
        }
        dst[i] = v;
    }
}

// ---------------------------------------------------------------------
// BN stats over [P][C] (C pow2): fixed grid so stride%C==0 -> each thread
// privately accumulates one channel, single LDS atomic, one global flush.
// ---------------------------------------------------------------------
__global__ __launch_bounds__(256)
void bn_stats_kernel(const float* __restrict__ t,
                     float* __restrict__ sums, float* __restrict__ sumsq,
                     long long total, int Cmask, int C)
{
    __shared__ float lsum[512];
    __shared__ float lsq[512];
    for (int c = threadIdx.x; c < C; c += 256) { lsum[c] = 0.f; lsq[c] = 0.f; }
    __syncthreads();
    long long i0     = (long long)blockIdx.x * 256 + threadIdx.x;
    long long stride = (long long)gridDim.x * 256;
    int c = (int)(i0 & Cmask);
    float s = 0.f, sq = 0.f;
    for (long long i = i0; i < total; i += stride) {
        float v = t[i];
        s += v;
        sq += v * v;
    }
    atomicAdd(&lsum[c], s);
    atomicAdd(&lsq[c], sq);
    __syncthreads();
    for (int c2 = threadIdx.x; c2 < C; c2 += 256) {
        atomicAdd(&sums[c2], lsum[c2]);
        atomicAdd(&sumsq[c2], lsq[c2]);
    }
}

__global__ void bn_finalize_kernel(const float* __restrict__ sums,
                                   const float* __restrict__ sumsq,
                                   const float* __restrict__ g,
                                   const float* __restrict__ b,
                                   float* __restrict__ scale,
                                   float* __restrict__ shift,
                                   float invN, int C)
{
    int c = blockIdx.x * blockDim.x + threadIdx.x;
    if (c < C) {
        float m = sums[c] * invN;
        float v = sumsq[c] * invN - m * m;
        float s = g[c] * rsqrtf(v + 1e-5f);
        scale[c] = s;
        shift[c] = b[c] - m * s;
    }
}

// BN+ReLU in place (f32 NHWC) — for double-conv outputs kept as skips
__global__ void bn_relu_kernel(float* __restrict__ t,
                               const float* __restrict__ scale,
                               const float* __restrict__ shift,
                               long long total, int Cmask)
{
    long long stride = (long long)gridDim.x * blockDim.x;
    for (long long i = (long long)blockIdx.x * blockDim.x + threadIdx.x; i < total;
         i += stride) {
        int c = (int)(i & Cmask);
        float v = t[i] * scale[c] + shift[c];
        t[i] = v > 0.f ? v : 0.f;
    }
}

// Fused BN+ReLU + zero-pad + f16 convert: raw conv1 out -> next conv's input
__global__ void bn_relu_pad_f16_kernel(const float* __restrict__ t,
                                       const float* __restrict__ scale,
                                       const float* __restrict__ shift,
                                       _Float16* __restrict__ dst,
                                       int Hh, int Ww, int C)
{
    int Hp = Hh + 2, Wp = Ww + 2;
    int total = 2 * Hp * Wp * C;
    for (int i = blockIdx.x * blockDim.x + threadIdx.x; i < total;
         i += gridDim.x * blockDim.x) {
        int c = i % C;
        int q = i / C;
        int x = q % Wp;
        int r = q / Wp;
        int y = r % Hp;
        int b = r / Hp;
        float v = 0.f;
        if (y > 0 && y < Hp - 1 && x > 0 && x < Wp - 1) {
            float raw = t[(((size_t)b * Hh + (y - 1)) * Ww + (x - 1)) * C + c];
            v = raw * scale[c] + shift[c];
            v = v > 0.f ? v : 0.f;
        }
        dst[i] = (_Float16)v;
    }
}

// ---------------------------------------------------------------------
// Scharr magnitude, reflect-101 border, NHWC f32 per-channel
// ---------------------------------------------------------------------
__global__ void scharr_kernel(const float* __restrict__ src, float* __restrict__ dst,
                              int Hh, int Ww, int C)
{
    int total = 2 * Hh * Ww * C;
    for (int i = blockIdx.x * blockDim.x + threadIdx.x; i < total;
         i += gridDim.x * blockDim.x) {
        int c = i % C;
        int q = i / C;
        int x = q % Ww;
        int r = q / Ww;
        int y = r % Hh;
        int b = r / Hh;
        const float* s = src + (size_t)b * Hh * Ww * C;
        int ym = (y == 0) ? 1 : y - 1;
        int yp = (y == Hh - 1) ? Hh - 2 : y + 1;
        int xm = (x == 0) ? 1 : x - 1;
        int xp = (x == Ww - 1) ? Ww - 2 : x + 1;
        #define AT(yy, xx) s[((size_t)(yy) * Ww + (xx)) * C + c]
        float a00 = AT(ym, xm), a01 = AT(ym, x), a02 = AT(ym, xp);
        float a10 = AT(y,  xm),                  a12 = AT(y,  xp);
        float a20 = AT(yp, xm), a21 = AT(yp, x), a22 = AT(yp, xp);
        #undef AT
        float gx = 3.f * (a02 - a00) + 10.f * (a12 - a10) + 3.f * (a22 - a20);
        float gy = 3.f * (a20 - a00) + 10.f * (a21 - a01) + 3.f * (a22 - a02);
        dst[i] = sqrtf(gx * gx + gy * gy);
    }
}

// Concat(A[Ca], B[Cb]) -> zero-padded f16 NHWC [2, H+2, W+2, Ca+Cb]
__global__ void pad_concat_f16_kernel(const float* __restrict__ a, int Ca,
                                      const float* __restrict__ bsrc, int Cb,
                                      _Float16* __restrict__ dst, int Hh, int Ww)
{
    int Ct = Ca + Cb;
    int Hp = Hh + 2, Wp = Ww + 2;
    int total = 2 * Hp * Wp * Ct;
    for (int i = blockIdx.x * blockDim.x + threadIdx.x; i < total;
         i += gridDim.x * blockDim.x) {
        int c = i % Ct;
        int q = i / Ct;
        int x = q % Wp;
        int r = q / Wp;
        int y = r % Hp;
        int b = r / Hp;
        float v = 0.f;
        if (y > 0 && y < Hp - 1 && x > 0 && x < Wp - 1) {
            int yy = y - 1, xx = x - 1;
            if (c < Ca)
                v = a[(((size_t)b * Hh + yy) * Ww + xx) * Ca + c];
            else
                v = bsrc[(((size_t)b * Hh + yy) * Ww + xx) * Cb + (c - Ca)];
        }
        dst[i] = (_Float16)v;
    }
}

__global__ void maxpool_kernel(const float* __restrict__ src, float* __restrict__ dst,
                               int Hh, int Ww, int C)
{
    int Ho = Hh >> 1, Wo = Ww >> 1;
    int total = 2 * Ho * Wo * C;
    for (int i = blockIdx.x * blockDim.x + threadIdx.x; i < total;
         i += gridDim.x * blockDim.x) {
        int c = i % C;
        int q = i / C;
        int x = q % Wo;
        int r = q / Wo;
        int y = r % Ho;
        int b = r / Ho;
        const float* s = src + (((size_t)b * Hh + 2 * y) * Ww + 2 * x) * C + c;
        size_t rs = (size_t)Ww * C;
        dst[i] = fmaxf(fmaxf(s[0], s[C]), fmaxf(s[rs], s[rs + C]));
    }
}

// bilinear 2x upsample, align_corners=True
__global__ void upsample2x_kernel(const float* __restrict__ src, float* __restrict__ dst,
                                  int Hh, int Ww, int C)
{
    int Ho = Hh * 2, Wo = Ww * 2;
    int total = 2 * Ho * Wo * C;
    float ry = (float)(Hh - 1) / (float)(Ho - 1);
    float rx = (float)(Ww - 1) / (float)(Wo - 1);
    for (int i = blockIdx.x * blockDim.x + threadIdx.x; i < total;
         i += gridDim.x * blockDim.x) {
        int c = i % C;
        int q = i / C;
        int x = q % Wo;
        int r = q / Wo;
        int y = r % Ho;
        int b = r / Ho;
        float fy = y * ry;
        int y0 = (int)fy;
        int y1 = y0 + 1; if (y1 > Hh - 1) y1 = Hh - 1;
        float wy = fy - (float)y0;
        float fx = x * rx;
        int x0 = (int)fx;
        int x1 = x0 + 1; if (x1 > Ww - 1) x1 = Ww - 1;
        float wx = fx - (float)x0;
        const float* s = src + (size_t)b * Hh * Ww * C + c;
        float v00 = s[((size_t)y0 * Ww + x0) * C];
        float v01 = s[((size_t)y0 * Ww + x1) * C];
        float v10 = s[((size_t)y1 * Ww + x0) * C];
        float v11 = s[((size_t)y1 * Ww + x1) * C];
        float top = v00 + (v01 - v00) * wx;
        float bot = v10 + (v11 - v10) * wx;
        dst[i] = top + (bot - top) * wy;
    }
}

// 1x1 head: y4 f32 NHWC [2,H,W,64] -> d_out f32 NCHW [2,2,H,W]
__global__ void out_conv_kernel(const float* __restrict__ y4, const float* __restrict__ w,
                                const float* __restrict__ bias, float* __restrict__ out,
                                int Hh, int Ww)
{
    int HWl = Hh * Ww;
    int total = 2 * HWl;
    int i = blockIdx.x * blockDim.x + threadIdx.x;
    if (i >= total) return;
    const float* p = y4 + (size_t)i * 64;
    float s0 = bias[0], s1 = bias[1];
    #pragma unroll 8
    for (int c = 0; c < 64; ++c) {
        float v = p[c];
        s0 += v * w[c];
        s1 += v * w[64 + c];
    }
    int b = i / HWl, q = i - b * HWl;
    out[((size_t)b * 2 + 0) * HWl + q] = s0;
    out[((size_t)b * 2 + 1) * HWl + q] = s1;
}

// =====================================================================
// Host orchestration
// =====================================================================
struct DCp { const float *w1, *g1, *b1, *w2, *g2, *b2; };

static inline int cdiv(int a, int b) { return (a + b - 1) / b; }

// convert weights, run conv, compute BN stats -> scale/shift in stat[2048/3072]
static void run_conv_raw(hipStream_t s, const _Float16* pin,
                         const float* w, const float* g, const float* bb,
                         _Float16* wcv, float* stat, float* outb,
                         int Hh, int Ww, int Cin, int Cout)
{
    int K = 9 * Cin, Kp = (K + 31) & ~31;
    int wtot = Cout * Kp;
    convert_weights_kernel<<<cdiv(wtot, 256), 256, 0, s>>>(w, wcv, Cout, Cin);
    int P = 2 * Hh * Ww;
    dim3 grid(P / TN, Cout / TM);
    conv3x3_wmma_kernel<<<grid, 256, 0, s>>>(pin, wcv, outb, Hh, Ww, Cin, Cout);
    hipMemsetAsync(stat, 0, 2048 * sizeof(float), s);
    long long total = (long long)P * Cout;
    bn_stats_kernel<<<512, 256, 0, s>>>(outb, stat, stat + 1024, total, Cout - 1, Cout);
    bn_finalize_kernel<<<cdiv(Cout, 64), 64, 0, s>>>(stat, stat + 1024, g, bb,
                                                     stat + 2048, stat + 3072,
                                                     1.0f / (float)P, Cout);
}

static void run_double_conv(hipStream_t s, _Float16* pinbuf, const DCp& p,
                            int Cin, int Cmid, int Cout, int Hh, int Ww,
                            float* tmid, float* outb, _Float16* wcv, float* stat)
{
    // conv1 -> raw mid; fused BN+ReLU+pad+f16 into conv2's input
    run_conv_raw(s, pinbuf, p.w1, p.g1, p.b1, wcv, stat, tmid, Hh, Ww, Cin, Cmid);
    int tot = 2 * (Hh + 2) * (Ww + 2) * Cmid;
    bn_relu_pad_f16_kernel<<<cdiv(tot, 256), 256, 0, s>>>(
        tmid, stat + 2048, stat + 3072, pinbuf, Hh, Ww, Cmid);
    // conv2 -> raw out; BN+ReLU in place (kept in f32 for skips/pool/upsample)
    run_conv_raw(s, pinbuf, p.w2, p.g2, p.b2, wcv, stat, outb, Hh, Ww, Cmid, Cout);
    long long total = (long long)2 * Hh * Ww * Cout;
    bn_relu_kernel<<<2048, 256, 0, s>>>(outb, stat + 2048, stat + 3072, total, Cout - 1);
}

extern "C" void kernel_launch(void* const* d_in, const int* in_sizes, int n_in,
                              void* d_out, int out_size, void* d_ws, size_t ws_size,
                              hipStream_t stream)
{
    (void)in_sizes; (void)n_in; (void)out_size; (void)ws_size;

    const float* x = (const float*)d_in[0];      // [2,1,512,512]; C=1 -> NHWC==NCHW
    DCp dc[9];
    int idx = 1;
    for (int i = 0; i < 9; ++i) {
        dc[i].w1 = (const float*)d_in[idx++];
        dc[i].g1 = (const float*)d_in[idx++];
        dc[i].b1 = (const float*)d_in[idx++];
        dc[i].w2 = (const float*)d_in[idx++];
        dc[i].g2 = (const float*)d_in[idx++];
        dc[i].b2 = (const float*)d_in[idx++];
    }
    const float* out_w = (const float*)d_in[idx++];  // [2,64,1,1]
    const float* out_b = (const float*)d_in[idx++];  // [2]

    // ---- workspace carve (bump allocator, 256B aligned) ----
    char* wsp = (char*)d_ws;
    auto carve = [&](size_t bytes) {
        char* p = wsp;
        wsp += (bytes + 255) & ~(size_t)255;
        return p;
    };
    const size_t NBIG = 2ull * 512 * 512 * 64;            // 33.55M elems
    float*    X1   = (float*)carve(NBIG * 4);
    float*    X2   = (float*)carve(2ull * 256 * 256 * 128 * 4);
    float*    X3   = (float*)carve(2ull * 128 * 128 * 256 * 4);
    float*    X4   = (float*)carve(2ull * 64 * 64 * 512 * 4);
    float*    X5   = (float*)carve(2ull * 32 * 32 * 512 * 4);
    float*    Y1   = (float*)carve(2ull * 64 * 64 * 256 * 4);
    float*    Y2   = (float*)carve(2ull * 128 * 128 * 128 * 4);
    float*    Y3   = (float*)carve(2ull * 256 * 256 * 64 * 4);
    float*    Y4   = (float*)carve(NBIG * 4);
    float*    TMID = (float*)carve(NBIG * 4);             // conv1 raw outputs
    float*    SBUF = (float*)carve(NBIG * 4);             // scharr scratch
    float*    PA   = (float*)carve(2ull * 256 * 256 * 64 * 4);
    float*    PB   = (float*)carve(2ull * 256 * 256 * 64 * 4);
    float*    TUP  = (float*)carve(NBIG * 4);             // upsample scratch
    _Float16* PIN  = (_Float16*)carve(2ull * 514 * 514 * 128 * 2);  // padded f16 in
    _Float16* WCV  = (_Float16*)carve(512ull * 9216 * 2);           // f16 weights
    float*    STAT = (float*)carve(4096 * 4);

    #define EW(kern, total, ...) \
        kern<<<cdiv((total), 256), 256, 0, stream>>>(__VA_ARGS__)

    // ---- stem: cat(x, Scharr(x)) ----
    EW(scharr_kernel, 2 * 512 * 512 * 1, x, SBUF, 512, 512, 1);
    EW(pad_concat_f16_kernel, 2 * 514 * 514 * 2, x, 1, SBUF, 1, PIN, 512, 512);
    run_double_conv(stream, PIN, dc[0], 2, 64, 64, 512, 512, TMID, X1, WCV, STAT);

    // ---- encoder ----
    EW(scharr_kernel, 2 * 512 * 512 * 64, X1, SBUF, 512, 512, 64);
    EW(maxpool_kernel, 2 * 256 * 256 * 64, X1, PA, 512, 512, 64);
    EW(maxpool_kernel, 2 * 256 * 256 * 64, SBUF, PB, 512, 512, 64);
    EW(pad_concat_f16_kernel, 2 * 258 * 258 * 128, PA, 64, PB, 64, PIN, 256, 256);
    run_double_conv(stream, PIN, dc[1], 128, 128, 128, 256, 256, TMID, X2, WCV, STAT);

    EW(maxpool_kernel, 2 * 128 * 128 * 128, X2, PA, 256, 256, 128);
    EW(pad_concat_f16_kernel, 2 * 130 * 130 * 128, PA, 128, nullptr, 0, PIN, 128, 128);
    run_double_conv(stream, PIN, dc[2], 128, 256, 256, 128, 128, TMID, X3, WCV, STAT);

    EW(maxpool_kernel, 2 * 64 * 64 * 256, X3, PA, 128, 128, 256);
    EW(pad_concat_f16_kernel, 2 * 66 * 66 * 256, PA, 256, nullptr, 0, PIN, 64, 64);
    run_double_conv(stream, PIN, dc[3], 256, 512, 512, 64, 64, TMID, X4, WCV, STAT);

    EW(maxpool_kernel, 2 * 32 * 32 * 512, X4, PA, 64, 64, 512);
    EW(pad_concat_f16_kernel, 2 * 34 * 34 * 512, PA, 512, nullptr, 0, PIN, 32, 32);
    run_double_conv(stream, PIN, dc[4], 512, 512, 512, 32, 32, TMID, X5, WCV, STAT);

    // ---- decoder (spatial dims match exactly -> no crop/pad delta) ----
    EW(upsample2x_kernel, 2 * 64 * 64 * 512, X5, TUP, 32, 32, 512);
    EW(pad_concat_f16_kernel, 2 * 66 * 66 * 1024, X4, 512, TUP, 512, PIN, 64, 64);
    run_double_conv(stream, PIN, dc[5], 1024, 512, 256, 64, 64, TMID, Y1, WCV, STAT);

    EW(upsample2x_kernel, 2 * 128 * 128 * 256, Y1, TUP, 64, 64, 256);
    EW(pad_concat_f16_kernel, 2 * 130 * 130 * 512, X3, 256, TUP, 256, PIN, 128, 128);
    run_double_conv(stream, PIN, dc[6], 512, 256, 128, 128, 128, TMID, Y2, WCV, STAT);

    EW(upsample2x_kernel, 2 * 256 * 256 * 128, Y2, TUP, 128, 128, 128);
    EW(pad_concat_f16_kernel, 2 * 258 * 258 * 256, X2, 128, TUP, 128, PIN, 256, 256);
    run_double_conv(stream, PIN, dc[7], 256, 128, 64, 256, 256, TMID, Y3, WCV, STAT);

    EW(upsample2x_kernel, 2 * 512 * 512 * 64, Y3, TUP, 256, 256, 64);
    EW(pad_concat_f16_kernel, 2 * 514 * 514 * 128, X1, 64, TUP, 64, PIN, 512, 512);
    run_double_conv(stream, PIN, dc[8], 128, 64, 64, 512, 512, TMID, Y4, WCV, STAT);

    // ---- 1x1 head -> NCHW f32 output ----
    EW(out_conv_kernel, 2 * 512 * 512, Y4, out_w, out_b, (float*)d_out, 512, 512);

    #undef EW
}
